// SSA_41489384079941
// MI455X (gfx1250) — matrix operand order repaired and checked by reading, not compiled, not run on hardware
//
#include <hip/hip_runtime.h>
#include <hip/hip_bf16.h>
#include <math.h>

typedef __attribute__((ext_vector_type(16))) __bf16 v16bf;
typedef __attribute__((ext_vector_type(8)))  float  v8f;
typedef __attribute__((ext_vector_type(4)))  unsigned int v4u;

#define BATCH   4
#define NTOK    65536           // tokens per batch (256*256)
#define CDIM    96
#define NHEADS  6
#define MROWS   (BATCH * NTOK)  // 262144 total tokens

__device__ __forceinline__ unsigned short f2bf(float f) {
  unsigned int u = __float_as_uint(f);
  u += 0x7FFFu + ((u >> 16) & 1u);      // round-to-nearest-even
  return (unsigned short)(u >> 16);
}
__device__ __forceinline__ float bf2f(unsigned short u) {
  return __uint_as_float(((unsigned int)u) << 16);
}
__device__ __forceinline__ unsigned int packbf(float lo, float hi) {
  return (unsigned int)f2bf(lo) | ((unsigned int)f2bf(hi) << 16);
}

// ---------------------------------------------------------------------------
// Weight pre-pack: W[K,N] fp32 -> bf16 in exact WMMA-B fragment order:
// PW[((nt16*NK + kb)*32 + lane)*16 + j] with k = kb*32 + (lane>>4)*16 + j,
// col = nt16*16 + (lane&15). One thread per output element.
// ---------------------------------------------------------------------------
__global__ void pack_w_bf16(const float* __restrict__ W,
                            unsigned short* __restrict__ PW,
                            int NK, int N) {
  int tid = blockIdx.x * 256 + threadIdx.x;
  int j    = tid & 15;
  int lane = (tid >> 4) & 31;
  int rest = tid >> 9;            // nt16*NK + kb
  int kb   = rest % NK;
  int nt16 = rest / NK;
  int k    = kb * 32 + (lane >> 4) * 16 + j;
  int col  = nt16 * 16 + (lane & 15);
  PW[tid] = f2bf(W[(size_t)k * N + col]);
}

// ---------------------------------------------------------------------------
// bf16 WMMA GEMM: out = A[M,K](bf16 row-major) @ PW (packed bf16)
// Compile-time flags: BIAS (add bias[N]), RES (add fp32 residual, fp32 out),
// OUTBF (bf16 output, no residual). Straight-line epilogue.
// block = 256 threads = 8 waves; wave: 16 rows x full N sweep.
// All B fragments of a tile loaded up-front (one clause) before the WMMA
// chain so loads overlap with matrix ops.
// ---------------------------------------------------------------------------
template<int K, bool BIAS, bool RES, bool OUTBF>
__global__ void gemm_bf16_wmma(const unsigned short* __restrict__ A,
                               const unsigned short* __restrict__ PW,
                               const float* __restrict__ bias,
                               const float* __restrict__ residual,
                               float* __restrict__ outf,
                               unsigned short* __restrict__ outb, int N) {
  constexpr int NK = K / 32;
  const int lane = threadIdx.x & 31;
  const int wave = threadIdx.x >> 5;
  const int row0 = (blockIdx.x * 8 + wave) * 16;
  const int half = lane >> 4;
  const int l16  = lane & 15;

  union Frag { v16bf v; v4u q[2]; };

  // A fragments: lane = row (l16), half-lane picks K sub-ranges; here k =
  // 16*half + s, one contiguous 16xbf16 run (two 8-wide loads), same order as PW.
  v16bf afrag[NK];
  const unsigned short* arow = A + (size_t)(row0 + l16) * K;
#pragma unroll
  for (int kb = 0; kb < NK; ++kb) {
    Frag fa;
    fa.q[0] = *(const v4u*)(arow + kb * 32 + half * 16);
    fa.q[1] = *(const v4u*)(arow + kb * 32 + half * 16 + 8);
    afrag[kb] = fa.v;
  }

  for (int nt = 0; nt < N; nt += 16) {
    const unsigned short* wbase =
        PW + ((size_t)(nt >> 4) * NK * 32 + lane) * 16;
    // load ALL B fragments of this tile first (single load clause)
    v16bf bfrag[NK];
#pragma unroll
    for (int kb = 0; kb < NK; ++kb) {
      Frag fb;
      const unsigned short* q = wbase + (size_t)kb * 32 * 16;
      fb.q[0] = ((const v4u*)q)[0];
      fb.q[1] = ((const v4u*)q)[1];
      bfrag[kb] = fb.v;
    }
    const int col = nt + l16;
    float bv = 0.f;
    if constexpr (BIAS) bv = bias[col];

    v8f acc = {0.f,0.f,0.f,0.f,0.f,0.f,0.f,0.f};
#pragma unroll
    for (int kb = 0; kb < NK; ++kb)
      acc = __builtin_amdgcn_wmma_f32_16x16x32_bf16(false, afrag[kb], false, bfrag[kb],
                                                    (short)0, acc, false, false);

    if constexpr (OUTBF) {
#pragma unroll
      for (int r = 0; r < 8; ++r) {
        const int m = row0 + half * 8 + r;
        outb[(size_t)m * N + col] = f2bf(acc[r] + bv);
      }
    } else {
#pragma unroll
      for (int r = 0; r < 8; ++r) {
        const int m = row0 + half * 8 + r;
        size_t idx = (size_t)m * N + col;
        float v = acc[r] + bv;
        if constexpr (RES) v += residual[idx];
        outf[idx] = v;
      }
    }
  }
}

// ---------------------------------------------------------------------------
// LayerNorm over C=96 (fp32 in, bf16 out), one wave per token
// ---------------------------------------------------------------------------
__global__ void layernorm_kernel(const float* __restrict__ x,
                                 const float* __restrict__ w,
                                 const float* __restrict__ b,
                                 unsigned short* __restrict__ out) {
  const int lane = threadIdx.x & 31;
  const int wave = threadIdx.x >> 5;
  const size_t tok = (size_t)blockIdx.x * 8 + wave;
  const float* xr = x + tok * CDIM;
  float v0 = xr[lane], v1 = xr[lane + 32], v2 = xr[lane + 64];
  float s = v0 + v1 + v2;
  for (int m = 16; m >= 1; m >>= 1) s += __shfl_xor(s, m, 32);
  float mu = s * (1.f / 96.f);
  float d0 = v0 - mu, d1 = v1 - mu, d2 = v2 - mu;
  float vs = d0 * d0 + d1 * d1 + d2 * d2;
  for (int m = 16; m >= 1; m >>= 1) vs += __shfl_xor(vs, m, 32);
  float inv = rsqrtf(vs * (1.f / 96.f) + 1e-5f);
  unsigned short* orow = out + tok * CDIM;
  orow[lane]      = f2bf(d0 * inv * w[lane]      + b[lane]);
  orow[lane + 32] = f2bf(d1 * inv * w[lane + 32] + b[lane + 32]);
  orow[lane + 64] = f2bf(d2 * inv * w[lane + 64] + b[lane + 64]);
}

// ---------------------------------------------------------------------------
// Window attention (bf16 q/kv in, bf16 out in win-reversed (B,N,C) layout)
// ---------------------------------------------------------------------------
__global__ void spa_attn_kernel(const unsigned short* __restrict__ Q,
                                const unsigned short* __restrict__ KV,
                                const float* __restrict__ rpb,
                                unsigned short* __restrict__ out) {
  __shared__ float kv_s[NHEADS][2][16][16];   // [head][k/v][pos][dh]
  const int wid = blockIdx.x;
  const int b  = wid >> 10;
  const int hb = (wid >> 5) & 31;
  const int wb = wid & 31;
  const int h    = threadIdx.x >> 5;
  const int lane = threadIdx.x & 31;
  const size_t base = (size_t)b * NTOK;

  for (int e = lane; e < 512; e += 32) {
    int t = e >> 8, idx = e & 255, p = idx >> 4, d = idx & 15;
    int flat = h * 16 + d;
    int si = flat / 48, sj = (flat % 48) / 24, c = flat % 24;
    int lr = (p >> 2) * 2 + si, lc = (p & 3) * 2 + sj;
    size_t gtok = base + (size_t)(hb * 8 + lr) * 256 + (wb * 8 + lc);
    kv_s[h][t][p][d] = bf2f(KV[gtok * 48 + t * 24 + c]);
  }
  __syncthreads();

  for (int qq = 0; qq < 2; ++qq) {
    int tl = lane + qq * 32;
    int r = tl >> 3, cc = tl & 7;
    size_t gtok = base + (size_t)(hb * 8 + r) * 256 + (wb * 8 + cc);
    const unsigned int* q32 = (const unsigned int*)(Q + gtok * 96 + h * 16);
    float qreg[16];
#pragma unroll
    for (int d2 = 0; d2 < 8; ++d2) {
      unsigned int u = q32[d2];
      qreg[2 * d2]     = bf2f((unsigned short)(u & 0xffffu));
      qreg[2 * d2 + 1] = bf2f((unsigned short)(u >> 16));
    }
    int qi = r >> 1, qj = cc >> 1;
    float logit[16], mx = -1e30f;
    for (int p = 0; p < 16; ++p) {
      float s = 0.f;
#pragma unroll
      for (int d = 0; d < 16; ++d) s += qreg[d] * kv_s[h][0][p][d];
      int bi = (qi - (p >> 2) + 3) * 7 + (qj - (p & 3) + 3);
      float l = s * 0.25f + rpb[bi * NHEADS + h];
      logit[p] = l; mx = fmaxf(mx, l);
    }
    float sum = 0.f;
    for (int p = 0; p < 16; ++p) { logit[p] = __expf(logit[p] - mx); sum += logit[p]; }
    float inv = 1.f / sum;
    float o[16];
    for (int d = 0; d < 16; ++d) {
      float s = 0.f;
#pragma unroll
      for (int p = 0; p < 16; ++p) s += logit[p] * kv_s[h][1][p][d];
      o[d] = s * inv;
    }
    unsigned int* ov = (unsigned int*)(out + gtok * 96 + h * 16);
#pragma unroll
    for (int d2 = 0; d2 < 8; ++d2) ov[d2] = packbf(o[2 * d2], o[2 * d2 + 1]);
  }
}

// ---------------------------------------------------------------------------
// Channel attention: 8x16 attn accumulation (bf16 q/kv in)
// ---------------------------------------------------------------------------
__global__ void sca_attn_acc(const unsigned short* __restrict__ Q,
                             const unsigned short* __restrict__ KV,
                             float* __restrict__ attn) {
  __shared__ float lacc[128];
  const int bh = blockIdx.x >> 8;
  const int b = bh / NHEADS, h = bh % NHEADS;
  const int n = ((blockIdx.x & 255) << 8) + threadIdx.x;
  if (threadIdx.x < 128) lacc[threadIdx.x] = 0.f;
  __syncthreads();
  const unsigned short* qb = Q  + (size_t)b * (NTOK * CDIM);
  const unsigned short* kb = KV + (size_t)b * (NTOK * CDIM);
  float q[16], k[8], qs = 0.f, ks = 0.f;
#pragma unroll
  for (int d = 0; d < 16; ++d) {
    q[d] = bf2f(qb[(size_t)h * 1048576 + (size_t)d * 65536 + n]); qs += q[d] * q[d];
  }
#pragma unroll
  for (int e = 0; e < 8; ++e) {
    size_t e2 = (size_t)h * 524288 + (size_t)e * 65536 + n;
    size_t tok = e2 / 48; int ch = (int)(e2 - tok * 48);
    k[e] = bf2f(kb[tok * 96 + ch]); ks += k[e] * k[e];
  }
  float qin = 1.f / fmaxf(sqrtf(qs), 1e-12f);
  float kin = 1.f / fmaxf(sqrtf(ks), 1e-12f);
#pragma unroll
  for (int e = 0; e < 8; ++e)
#pragma unroll
    for (int d = 0; d < 16; ++d)
      atomicAdd(&lacc[e * 16 + d], (k[e] * kin) * (q[d] * qin));
  __syncthreads();
  if (threadIdx.x < 128) atomicAdd(&attn[bh * 128 + threadIdx.x], lacc[threadIdx.x]);
}

__global__ void sca_softmax_kernel(float* __restrict__ attn) {
  float* row = attn + threadIdx.x * 16;   // 192 rows = 24 (b,h) * 8 e
  float mx = -1e30f;
  for (int d = 0; d < 16; ++d) mx = fmaxf(mx, row[d]);
  float s = 0.f;
  for (int d = 0; d < 16; ++d) { row[d] = __expf(row[d] - mx); s += row[d]; }
  float inv = 1.f / s;
  for (int d = 0; d < 16; ++d) row[d] *= inv;
}

__global__ void sca_out_kernel(const unsigned short* __restrict__ KV,
                               const float* __restrict__ attn,
                               unsigned short* __restrict__ out) {
  __shared__ float att_s[128];
  const int bh = blockIdx.x >> 8;
  const int b = bh / NHEADS, h = bh % NHEADS;
  const int n = ((blockIdx.x & 255) << 8) + threadIdx.x;
  if (threadIdx.x < 128) att_s[threadIdx.x] = attn[bh * 128 + threadIdx.x];
  __syncthreads();
  const unsigned short* kvb = KV + (size_t)b * (NTOK * CDIM);
  float v[8], vs = 0.f;
#pragma unroll
  for (int e = 0; e < 8; ++e) {
    size_t e2 = (size_t)h * 524288 + (size_t)e * 65536 + n;
    size_t tok = e2 / 48; int ch = (int)(e2 - tok * 48) + 48;
    v[e] = bf2f(kvb[tok * 96 + ch]); vs += v[e] * v[e];
  }
  float vin = 1.f / fmaxf(sqrtf(vs), 1e-12f);
  float o[16];
#pragma unroll
  for (int d = 0; d < 16; ++d) {
    float s = 0.f;
#pragma unroll
    for (int e = 0; e < 8; ++e) s += v[e] * vin * att_s[e * 16 + d];
    o[d] = s;
  }
  unsigned int* ob = (unsigned int*)(out + (size_t)b * (NTOK * CDIM) +
                                     (size_t)h * 1048576 + (size_t)n * 16);
#pragma unroll
  for (int d2 = 0; d2 < 8; ++d2) ob[d2] = packbf(o[2 * d2], o[2 * d2 + 1]);
}

// ---------------------------------------------------------------------------
// Depthwise 3x3 (SAME) + bias + exact GELU * gate (bf16 H in, bf16 act out)
// ---------------------------------------------------------------------------
__global__ void dwconv_gelu_kernel(const unsigned short* __restrict__ Hb,
                                   const float* __restrict__ dww,
                                   const float* __restrict__ dwb,
                                   unsigned short* __restrict__ act) {
  size_t tid = (size_t)blockIdx.x * blockDim.x + threadIdx.x;
  int ch = (int)(tid & 255);
  size_t rest = tid >> 8;
  int n = (int)(rest & 65535);
  int b = (int)(rest >> 16);
  int y = n >> 8, x = n & 255;
  float s = 0.f;
#pragma unroll
  for (int dy = -1; dy <= 1; ++dy) {
    int yy = y + dy; if (yy < 0 || yy > 255) continue;
#pragma unroll
    for (int dx = -1; dx <= 1; ++dx) {
      int xx = x + dx; if (xx < 0 || xx > 255) continue;
      float a = bf2f(Hb[((size_t)b * 65536 + (size_t)yy * 256 + xx) * 512 + ch]);
      s += a * dww[ch * 9 + (dy + 1) * 3 + (dx + 1)];
    }
  }
  s += dwb[ch];
  float g = 0.5f * s * (1.f + erff(s * 0.70710678118654752f));
  float gate = bf2f(Hb[((size_t)b * 65536 + n) * 512 + 256 + ch]);
  act[((size_t)b * 65536 + n) * 256 + ch] = f2bf(g * gate);
}

// ---------------------------------------------------------------------------
extern "C" void kernel_launch(void* const* d_in, const int* in_sizes, int n_in,
                              void* d_out, int out_size, void* d_ws, size_t ws_size,
                              hipStream_t stream) {
  (void)in_sizes; (void)n_in; (void)out_size; (void)ws_size;
  const float* x       = (const float*)d_in[0];
  const float* n1w     = (const float*)d_in[1];
  const float* n1b     = (const float*)d_in[2];
  const float* spa_qw  = (const float*)d_in[3];
  const float* spa_qb  = (const float*)d_in[4];
  const float* spa_kvw = (const float*)d_in[5];
  const float* spa_kvb = (const float*)d_in[6];
  const float* spa_pw  = (const float*)d_in[7];
  const float* spa_pb  = (const float*)d_in[8];
  const float* rpb     = (const float*)d_in[9];
  const float* sca_qw  = (const float*)d_in[10];
  const float* sca_kvw = (const float*)d_in[11];
  const float* sca_pw  = (const float*)d_in[12];
  const float* n2w     = (const float*)d_in[13];
  const float* n2b     = (const float*)d_in[14];
  const float* fc1_w   = (const float*)d_in[15];
  const float* fc1_b   = (const float*)d_in[16];
  const float* dw_w    = (const float*)d_in[17];
  const float* dw_b    = (const float*)d_in[18];
  const float* fc2_w   = (const float*)d_in[19];
  const float* fc2_b   = (const float*)d_in[20];
  float* out = (float*)d_out;

  // ---- workspace carve-up (all regions 32B aligned) ----
  char* p = (char*)d_ws;
  unsigned short* XN  = (unsigned short*)p; p += (size_t)MROWS * CDIM * 2;  // ln out / sca out / ln2 out
  unsigned short* QS  = (unsigned short*)p; p += (size_t)MROWS * CDIM * 2;  // q (spa, then sca)
  unsigned short* KVS = (unsigned short*)p; p += (size_t)MROWS * 48 * 2;    // spa kv
  unsigned short* AT  = (unsigned short*)p; p += (size_t)MROWS * CDIM * 2;  // window-attn out
  float*          XA  = (float*)p;          p += (size_t)MROWS * CDIM * 4;  // running residual (fp32)
  unsigned short* KV2 = (unsigned short*)p; p += (size_t)MROWS * CDIM * 2;  // sca kv
  float*          SAT = (float*)p;          p += 24 * 128 * 4;              // sca attn acc
  unsigned short* PW  = (unsigned short*)p; p += 124416 * 2;                // packed weights
  unsigned short* HB  = (unsigned short*)p; p += (size_t)MROWS * 512 * 2;   // fc1 out
  unsigned short* ACT = (unsigned short*)p; p += (size_t)MROWS * 256 * 2;   // gated conv act

  // packed-weight offsets (ushort units)
  unsigned short* PW_spa_q  = PW;            // 96x96  -> 9216
  unsigned short* PW_spa_kv = PW + 9216;     // 96x48  -> 4608
  unsigned short* PW_spa_p  = PW + 13824;    // 96x96
  unsigned short* PW_sca_q  = PW + 23040;    // 96x96
  unsigned short* PW_sca_kv = PW + 32256;    // 96x96
  unsigned short* PW_sca_p  = PW + 41472;    // 96x96
  unsigned short* PW_fc1    = PW + 50688;    // 96x512 -> 49152
  unsigned short* PW_fc2    = PW + 99840;    // 256x96 -> 24576

  const int GB = MROWS / 128;       // 2048 blocks: 8 waves * 16 rows each

  // ---- pre-pack weights to bf16 fragment order (cheap; ~0.5 MB total) ----
  pack_w_bf16<<<(96 * 96) / 256, 256, 0, stream>>>(spa_qw,  PW_spa_q,  3, 96);
  pack_w_bf16<<<(96 * 48) / 256, 256, 0, stream>>>(spa_kvw, PW_spa_kv, 3, 48);
  pack_w_bf16<<<(96 * 96) / 256, 256, 0, stream>>>(spa_pw,  PW_spa_p,  3, 96);
  pack_w_bf16<<<(96 * 96) / 256, 256, 0, stream>>>(sca_qw,  PW_sca_q,  3, 96);
  pack_w_bf16<<<(96 * 96) / 256, 256, 0, stream>>>(sca_kvw, PW_sca_kv, 3, 96);
  pack_w_bf16<<<(96 * 96) / 256, 256, 0, stream>>>(sca_pw,  PW_sca_p,  3, 96);
  pack_w_bf16<<<(96 * 512) / 256, 256, 0, stream>>>(fc1_w,  PW_fc1,    3, 512);
  pack_w_bf16<<<(256 * 96) / 256, 256, 0, stream>>>(fc2_w,  PW_fc2,    8, 96);

  // ---- x -> ln1 (bf16) ----
  layernorm_kernel<<<MROWS / 8, 256, 0, stream>>>(x, n1w, n1b, XN);
  // ---- window attention branch ----
  gemm_bf16_wmma<96, true, false, true><<<GB, 256, 0, stream>>>(
      XN, PW_spa_q,  spa_qb,  nullptr, nullptr, QS,  96);
  gemm_bf16_wmma<96, true, false, true><<<GB, 256, 0, stream>>>(
      XN, PW_spa_kv, spa_kvb, nullptr, nullptr, KVS, 48);
  spa_attn_kernel<<<4096, 192, 0, stream>>>(QS, KVS, rpb, AT);
  gemm_bf16_wmma<96, true, true, false><<<GB, 256, 0, stream>>>(
      AT, PW_spa_p, spa_pb, x, XA, nullptr, 96);                       // + x
  // ---- channel attention branch ----
  gemm_bf16_wmma<96, false, false, true><<<GB, 256, 0, stream>>>(
      XN, PW_sca_q,  nullptr, nullptr, nullptr, QS,  96);
  gemm_bf16_wmma<96, false, false, true><<<GB, 256, 0, stream>>>(
      XN, PW_sca_kv, nullptr, nullptr, nullptr, KV2, 96);
  hipMemsetAsync(SAT, 0, 24 * 128 * sizeof(float), stream);
  sca_attn_acc<<<24 * 256, 256, 0, stream>>>(QS, KV2, SAT);
  sca_softmax_kernel<<<1, 192, 0, stream>>>(SAT);
  sca_out_kernel<<<24 * 256, 256, 0, stream>>>(KV2, SAT, XN);
  gemm_bf16_wmma<96, false, true, false><<<GB, 256, 0, stream>>>(
      XN, PW_sca_p, nullptr, XA, XA, nullptr, 96);                     // + residual
  // ---- FFN ----
  layernorm_kernel<<<MROWS / 8, 256, 0, stream>>>(XA, n2w, n2b, XN);
  gemm_bf16_wmma<96, true, false, true><<<GB, 256, 0, stream>>>(
      XN, PW_fc1, fc1_b, nullptr, nullptr, HB, 512);
  dwconv_gelu_kernel<<<MROWS, 256, 0, stream>>>(HB, dw_w, dw_b, ACT);
  gemm_bf16_wmma<256, true, true, false><<<GB, 256, 0, stream>>>(
      ACT, PW_fc2, fc2_b, XA, out, nullptr, 96);                       // + residual
}
